// MoEPatchEncoder_71605694759013
// MI455X (gfx1250) — matrix operand order repaired and checked by reading, not compiled, not run on hardware
//
#include <hip/hip_runtime.h>
#include <hip/hip_bf16.h>
#include <math.h>

typedef _Float16 half_t;
typedef __attribute__((ext_vector_type(16))) _Float16 v16h;
typedef __attribute__((ext_vector_type(8)))  _Float16 v8h;
typedef __attribute__((ext_vector_type(8)))  float    v8f;

#define WMMA_F16(a, b, c) \
  __builtin_amdgcn_wmma_f32_16x16x32_f16(false, (a), false, (b), (short)0, (c), false, false)

// ---------------------------------------------------------------------------
// Fragment loaders (wave32): A and B share one contiguous K order (k = 16*(l>>4) + s)
// A (16x32 f16): lane l -> M = l&15; 16 contiguous K at k0 + (l>=16?16:0)
// B (32x16 f16) from W[N][K] row-major (out = A @ W^T):
//   lane l -> N = n0 + (l&15); 16 contiguous K at k0 + (l>=16?16:0)
// C/D (16x16 f32): vgpr r, lane l -> M = r + 8*(l>>4), N = l&15
// ---------------------------------------------------------------------------
__device__ __forceinline__ v16h load_a_frag(const half_t* rowptr, int k0) {
  const int lane = threadIdx.x & 31;
  const int koff = k0 + ((lane >> 4) << 4);
  v8h lo = *(const v8h*)(rowptr + koff);
  v8h hi = *(const v8h*)(rowptr + koff + 8);
  v16h a;
#pragma unroll
  for (int i = 0; i < 8; ++i) { a[i] = lo[i]; a[i + 8] = hi[i]; }
  return a;
}

__device__ __forceinline__ v16h load_b_frag_f32(const float* W, long ldk, int n0, int k0) {
  const int lane = threadIdx.x & 31;
  const float* p = W + (long)(n0 + (lane & 15)) * ldk + (k0 + ((lane >> 4) << 4));
  v16h b;
#pragma unroll
  for (int i = 0; i < 16; ++i) b[i] = (half_t)p[i];
  return b;
}

__device__ __forceinline__ v16h load_b_frag_f16(const half_t* W, long ldk, int n0, int k0) {
  const int lane = threadIdx.x & 31;
  const half_t* p = W + (long)(n0 + (lane & 15)) * ldk + (k0 + ((lane >> 4) << 4));
  v8h lo = *(const v8h*)p;
  v8h hi = *(const v8h*)(p + 8);
  v16h b;
#pragma unroll
  for (int i = 0; i < 8; ++i) { b[i] = lo[i]; b[i + 8] = hi[i]; }
  return b;
}

// ---------------------------------------------------------------------------
// CDNA5 async global->LDS copy (ASYNCcnt-tracked, no VGPR data movement)
// ---------------------------------------------------------------------------
__device__ __forceinline__ void async_copy_b128_to_lds(void* lds_dst, const void* gsrc) {
  unsigned ldsa = (unsigned)(unsigned long long)lds_dst;       // LDS aperture: low 32 bits
  unsigned long long ga = (unsigned long long)gsrc;
  asm volatile("global_load_async_to_lds_b128 %0, %1, off"
               :: "v"(ldsa), "v"(ga) : "memory");
}
__device__ __forceinline__ void wait_asynccnt0() {
  asm volatile("s_wait_asynccnt 0x0" ::: "memory");
}

// ---------------------------------------------------------------------------
// Kernel: f32 -> f16 weight conversion (one streamed pass)
// ---------------------------------------------------------------------------
__global__ __launch_bounds__(256) void k_cvt_f32_f16(const float* __restrict__ src,
                                                     half_t* __restrict__ dst, long n8) {
  long i = (long)blockIdx.x * blockDim.x + threadIdx.x;
  const long stride = (long)gridDim.x * blockDim.x;
  for (; i < n8; i += stride) {
    v8f v = *(const v8f*)(src + i * 8);
    v8h h;
#pragma unroll
    for (int j = 0; j < 8; ++j) h[j] = (half_t)v[j];
    *(v8h*)(dst + i * 8) = h;
  }
}

// ---------------------------------------------------------------------------
// Kernel 1: unfold image -> feat (f32 for router, f16 for WMMA A operands)
// ---------------------------------------------------------------------------
__global__ __launch_bounds__(256) void k_unfold(const float* __restrict__ img,
                                                float* __restrict__ feat32,
                                                half_t* __restrict__ feat16) {
  const int p = blockIdx.x;
  const int gy = p >> 5, gx = p & 31;
#pragma unroll
  for (int j = 0; j < 3; ++j) {
    int q = threadIdx.x + 256 * j;     // 0..767 = c*256 + py*16 + px
    int c = q >> 8, rem = q & 255;
    int py = rem >> 4, px = rem & 15;
    float v = img[((c << 9) + gy * 16 + py) * 512 + (gx * 16 + px)];
    feat32[(long)p * 768 + q] = v;
    feat16[(long)p * 768 + q] = (half_t)v;
  }
}

// ---------------------------------------------------------------------------
// Kernel 2: router
// ---------------------------------------------------------------------------
__global__ __launch_bounds__(256) void k_router(const float* __restrict__ feat32,
                                                const float* __restrict__ rw1,
                                                const float* __restrict__ rb1,
                                                const float* __restrict__ rw2,
                                                const float* __restrict__ rb2,
                                                float* __restrict__ out_probs,
                                                float* __restrict__ out_eid_f,
                                                int* __restrict__ eid) {
  __shared__ float sf[768];
  __shared__ float shr[256];
  __shared__ float sred[256];
  __shared__ float slog[4];
  const int p = blockIdx.x;
  const int t = threadIdx.x;
  for (int i = t; i < 768; i += 256) sf[i] = feat32[(long)p * 768 + i];
  __syncthreads();
  float acc = 0.f;
  const float* wr = rw1 + (long)t * 768;
  for (int k = 0; k < 768; ++k) acc += wr[k] * sf[k];
  shr[t] = fmaxf(acc + rb1[t], 0.f);
  __syncthreads();
  for (int e = 0; e < 4; ++e) {
    sred[t] = shr[t] * rw2[e * 256 + t];
    __syncthreads();
    for (int s = 128; s > 0; s >>= 1) {
      if (t < s) sred[t] += sred[t + s];
      __syncthreads();
    }
    if (t == 0) slog[e] = sred[0] + rb2[e];
    __syncthreads();
  }
  if (t == 0) {
    float m = slog[0]; int arg = 0;
    for (int e = 1; e < 4; ++e) if (slog[e] > m) { m = slog[e]; arg = e; }
    float ex[4], s = 0.f;
    for (int e = 0; e < 4; ++e) { ex[e] = __expf(slog[e] - m); s += ex[e]; }
    for (int e = 0; e < 4; ++e) out_probs[p * 4 + e] = ex[e] / s;
    out_eid_f[p] = (float)arg;
    eid[p] = arg;
  }
}

// ---------------------------------------------------------------------------
// Kernel 3: deterministic per-expert compaction
// ---------------------------------------------------------------------------
__global__ __launch_bounds__(1024) void k_compact(const int* __restrict__ eid,
                                                  int* __restrict__ counts,
                                                  int* __restrict__ lists) {
  __shared__ int se[1024];
  const int t = threadIdx.x;
  se[t] = eid[t];
  __syncthreads();
  if (t < 4) {
    int c = 0;
    for (int i = 0; i < 1024; ++i)
      if (se[i] == t) lists[t * 1024 + (c++)] = i;
    counts[t] = c;
  }
}

__global__ __launch_bounds__(256) void k_zero(float* __restrict__ dst, long n) {
  long i = (long)blockIdx.x * blockDim.x + threadIdx.x;
  const long stride = (long)gridDim.x * blockDim.x;
  for (; i < n; i += stride) dst[i] = 0.f;
}

// ---------------------------------------------------------------------------
// Kernel 4: expert front-end on 16 gathered rows (x, v, attn, LayerNorm)
// ---------------------------------------------------------------------------
__global__ __launch_bounds__(256) void k_frontend(
    const half_t* __restrict__ feat16,
    const int* __restrict__ counts, const int* __restrict__ lists,
    const float* __restrict__ pe_w, const float* __restrict__ pe_b,
    const float* __restrict__ inw, const float* __restrict__ inb,
    const float* __restrict__ ow,  const float* __restrict__ ob,
    const float* __restrict__ ln_g, const float* __restrict__ ln_b,
    half_t* __restrict__ h_out) {
  const int e = blockIdx.x;
  const int tbase = blockIdx.y * 16;
  const int cnt = counts[e];
  if (tbase >= cnt) return;                       // block-uniform

  __shared__ __align__(16) half_t sfeat[16][768];
  __shared__ __align__(16) half_t sXh[16][256];
  __shared__ __align__(16) half_t sVh[16][256];
  __shared__ __align__(16) float  sR[16][256];
  __shared__ float smu[16], srs[16];
  __shared__ int srow[16];

  const int t = threadIdx.x;
  if (t < 16) {
    int gi = tbase + t; if (gi >= cnt) gi = cnt - 1;
    srow[t] = lists[e * 1024 + gi];
  }
  __syncthreads();
  // stage 16x768 f16 tile via async global->LDS (16B per thread per iter)
  for (int c = t; c < 16 * 96; c += 256) {        // 96 chunks of 8 halves per row
    int r = c / 96, cc = c % 96;
    async_copy_b128_to_lds(&sfeat[r][cc * 8], feat16 + (long)srow[r] * 768 + cc * 8);
  }
  wait_asynccnt0();
  __syncthreads();

  const int wave = t >> 5, lane = t & 31;
  const int mrow = lane & 15, mhi = lane >> 4;

  // --- x = feat @ pe_w^T + pe_b (K=768) ---
  const float* Wx = pe_w + (long)e * 256 * 768;
#pragma unroll
  for (int j = 0; j < 2; ++j) {
    const int n0 = (wave * 2 + j) * 16;
    v8f acc = {};
    for (int k0 = 0; k0 < 768; k0 += 32) {
      v16h A = load_a_frag(&sfeat[mrow][0], k0);
      v16h B = load_b_frag_f32(Wx, 768, n0, k0);
      acc = WMMA_F16(A, B, acc);
    }
    const int n = n0 + mrow;
    const float bias = pe_b[e * 256 + n];
#pragma unroll
    for (int r = 0; r < 8; ++r) {
      const int m = r + 8 * mhi;
      const float v = acc[r] + bias;
      sXh[m][n] = (half_t)v;
      sR[m][n]  = v;
    }
  }
  __syncthreads();

  // --- v = x @ Wv^T + bv ---
  const float* Wv = inw + ((long)e * 768 + 512) * 256;
#pragma unroll
  for (int j = 0; j < 2; ++j) {
    const int n0 = (wave * 2 + j) * 16;
    v8f acc = {};
    for (int k0 = 0; k0 < 256; k0 += 32) {
      v16h A = load_a_frag(&sXh[mrow][0], k0);
      v16h B = load_b_frag_f32(Wv, 256, n0, k0);
      acc = WMMA_F16(A, B, acc);
    }
    const int n = n0 + mrow;
    const float bias = inb[e * 768 + 512 + n];
#pragma unroll
    for (int r = 0; r < 8; ++r) sVh[r + 8 * mhi][n] = (half_t)(acc[r] + bias);
  }
  __syncthreads();

  // --- resid = x + (v @ Wo^T + ob) ---
  const float* Wo = ow + (long)e * 256 * 256;
#pragma unroll
  for (int j = 0; j < 2; ++j) {
    const int n0 = (wave * 2 + j) * 16;
    v8f acc = {};
    for (int k0 = 0; k0 < 256; k0 += 32) {
      v16h A = load_a_frag(&sVh[mrow][0], k0);
      v16h B = load_b_frag_f32(Wo, 256, n0, k0);
      acc = WMMA_F16(A, B, acc);
    }
    const int n = n0 + mrow;
    const float bias = ob[e * 256 + n];
#pragma unroll
    for (int r = 0; r < 8; ++r) sR[r + 8 * mhi][n] += acc[r] + bias;
  }
  __syncthreads();

  // --- LayerNorm per row ---
  if (t < 16) {
    float mu = 0.f;
    for (int c = 0; c < 256; ++c) mu += sR[t][c];
    mu *= (1.f / 256.f);
    float var = 0.f;
    for (int c = 0; c < 256; ++c) { float d = sR[t][c] - mu; var += d * d; }
    var *= (1.f / 256.f);
    smu[t] = mu;
    srs[t] = rsqrtf(var + 1e-5f);
  }
  __syncthreads();
  for (int i = t; i < 16 * 256; i += 256) {
    const int r = i >> 8, c = i & 255;
    if (tbase + r < cnt) {
      float v = (sR[r][c] - smu[r]) * srs[r] * ln_g[e * 256 + c] + ln_b[e * 256 + c];
      h_out[((long)e * 1024 + tbase + r) * 256 + c] = (half_t)v;
    }
  }
}

// ---------------------------------------------------------------------------
// Kernel 5: l1[patch] = relu(h @ p1_w[e]^T + p1_b) (M-tile 64, K=256, N=16384)
// ---------------------------------------------------------------------------
template <bool WH>
__global__ __launch_bounds__(256) void k_p1(
    const half_t* __restrict__ h_in,
    const int* __restrict__ counts, const int* __restrict__ lists,
    const float* __restrict__ p1_wf, const half_t* __restrict__ p1_wh,
    const float* __restrict__ p1_b, half_t* __restrict__ l1) {
  const int e = blockIdx.x;
  const int tbase = blockIdx.y * 64;
  const int cnt = counts[e];
  if (tbase >= cnt) return;

  __shared__ __align__(16) half_t sH[64][256];   // 32 KB
  __shared__ int spatch[64];
  const int t = threadIdx.x;
  if (t < 64) {
    int gi = tbase + t; if (gi >= cnt) gi = cnt - 1;
    spatch[t] = lists[e * 1024 + gi];
  }
  __syncthreads();
  // stage 64x256 f16 tile via async global->LDS (ASYNCcnt path)
  for (int c = t; c < 64 * 32; c += 256) {       // 32 chunks of 8 halves per row
    int r = c >> 5, cc = c & 31;
    int gi = tbase + r; if (gi >= cnt) gi = cnt - 1;
    async_copy_b128_to_lds(&sH[r][cc * 8],
                           h_in + ((long)e * 1024 + gi) * 256 + cc * 8);
  }
  wait_asynccnt0();
  __syncthreads();

  const int wave = t >> 5, lane = t & 31;
  const int mrow = lane & 15, mhi = lane >> 4;
  const float*  Wf = p1_wf + (long)e * 16384 * 256;
  const half_t* Wh = p1_wh + (long)e * 16384 * 256;
  const float*  Bv = p1_b + (long)e * 16384;

  for (int nt = wave; nt < 1024; nt += 8) {      // wave-uniform
    const int n0 = nt * 16;
    v8f acc0 = {}, acc1 = {}, acc2 = {}, acc3 = {};
#pragma unroll
    for (int k0 = 0; k0 < 256; k0 += 32) {
      v16h B;
      if constexpr (WH) B = load_b_frag_f16(Wh, 256, n0, k0);
      else              B = load_b_frag_f32(Wf, 256, n0, k0);
      v16h A0 = load_a_frag(&sH[ 0 + mrow][0], k0);
      v16h A1 = load_a_frag(&sH[16 + mrow][0], k0);
      v16h A2 = load_a_frag(&sH[32 + mrow][0], k0);
      v16h A3 = load_a_frag(&sH[48 + mrow][0], k0);
      acc0 = WMMA_F16(A0, B, acc0);
      acc1 = WMMA_F16(A1, B, acc1);
      acc2 = WMMA_F16(A2, B, acc2);
      acc3 = WMMA_F16(A3, B, acc3);
    }
    const int n = n0 + mrow;
    const float bias = Bv[n];
#pragma unroll
    for (int mt = 0; mt < 4; ++mt) {
      const v8f& acc = mt == 0 ? acc0 : mt == 1 ? acc1 : mt == 2 ? acc2 : acc3;
#pragma unroll
      for (int r = 0; r < 8; ++r) {
        const int mloc = mt * 16 + r + 8 * mhi;
        if (tbase + mloc < cnt) {
          const float v = fmaxf(acc[r] + bias, 0.f);
          l1[(long)spatch[mloc] * 16384 + n] = (half_t)v;
        }
      }
    }
  }
}

// ---------------------------------------------------------------------------
// Kernel 6: lat = tanh(l1 @ p2_w[e]^T + p2_b) scattered into masked output
// ---------------------------------------------------------------------------
template <bool WH>
__global__ __launch_bounds__(256) void k_p2(
    const half_t* __restrict__ l1,
    const int* __restrict__ counts, const int* __restrict__ lists,
    const float* __restrict__ w0, const float* __restrict__ w1,
    const float* __restrict__ w2, const float* __restrict__ w3,
    const half_t* __restrict__ wh_base,
    const float* __restrict__ b0, const float* __restrict__ b1,
    const float* __restrict__ b2, const float* __restrict__ b3,
    float* __restrict__ out_lat) {
  const int e = blockIdx.x;
  const int tbase = blockIdx.y * 64;
  const int cnt = counts[e];
  if (tbase >= cnt) return;

  const float* Wf; const float* Bv; int Nc, coff; long whoff;
  if      (e == 0) { Wf = w0; Bv = b0; Nc = 128;  coff = 0;   whoff = 0L;        }
  else if (e == 1) { Wf = w1; Bv = b1; Nc = 256;  coff = 128; whoff = 2097152L;  }
  else if (e == 2) { Wf = w2; Bv = b2; Nc = 512;  coff = 384; whoff = 6291456L;  }
  else             { Wf = w3; Bv = b3; Nc = 1024; coff = 896; whoff = 14680064L; }
  const half_t* Wh = wh_base + whoff;

  __shared__ int spatch[64];
  const int t = threadIdx.x;
  if (t < 64) {
    int gi = tbase + t; if (gi >= cnt) gi = cnt - 1;
    spatch[t] = lists[e * 1024 + gi];
  }
  __syncthreads();

  const int wave = t >> 5, lane = t & 31;
  const int mrow = lane & 15, mhi = lane >> 4;
  const half_t* a0 = l1 + (long)spatch[ 0 + mrow] * 16384;
  const half_t* a1 = l1 + (long)spatch[16 + mrow] * 16384;
  const half_t* a2 = l1 + (long)spatch[32 + mrow] * 16384;
  const half_t* a3 = l1 + (long)spatch[48 + mrow] * 16384;

  const int ntiles = Nc >> 4;
  for (int nt = wave; nt < ntiles; nt += 8) {    // wave-uniform
    const int n0 = nt * 16;
    v8f acc0 = {}, acc1 = {}, acc2 = {}, acc3 = {};
#pragma unroll 4
    for (int k0 = 0; k0 < 16384; k0 += 32) {
      v16h B;
      if constexpr (WH) B = load_b_frag_f16(Wh, 16384, n0, k0);
      else              B = load_b_frag_f32(Wf, 16384, n0, k0);
      acc0 = WMMA_F16(load_a_frag(a0, k0), B, acc0);
      acc1 = WMMA_F16(load_a_frag(a1, k0), B, acc1);
      acc2 = WMMA_F16(load_a_frag(a2, k0), B, acc2);
      acc3 = WMMA_F16(load_a_frag(a3, k0), B, acc3);
    }
    const int n = n0 + mrow;
    const float bias = Bv[n];
#pragma unroll
    for (int mt = 0; mt < 4; ++mt) {
      const v8f& acc = mt == 0 ? acc0 : mt == 1 ? acc1 : mt == 2 ? acc2 : acc3;
#pragma unroll
      for (int r = 0; r < 8; ++r) {
        const int mloc = mt * 16 + r + 8 * mhi;
        if (tbase + mloc < cnt) {
          out_lat[(long)spatch[mloc] * 1920 + coff + n] = tanhf(acc[r] + bias);
        }
      }
    }
  }
}

// ---------------------------------------------------------------------------
extern "C" void kernel_launch(void* const* d_in, const int* in_sizes, int n_in,
                              void* d_out, int out_size, void* d_ws, size_t ws_size,
                              hipStream_t stream) {
  (void)in_sizes; (void)n_in; (void)out_size;
  const float* image = (const float*)d_in[0];
  const float* rw1   = (const float*)d_in[1];
  const float* rb1   = (const float*)d_in[2];
  const float* rw2   = (const float*)d_in[3];
  const float* rb2   = (const float*)d_in[4];
  const float* pe_w  = (const float*)d_in[5];
  const float* pe_b  = (const float*)d_in[6];
  const float* inw   = (const float*)d_in[7];
  const float* inb   = (const float*)d_in[8];
  const float* oww   = (const float*)d_in[9];
  const float* ob    = (const float*)d_in[10];
  const float* ln_g  = (const float*)d_in[11];
  const float* ln_b  = (const float*)d_in[12];
  const float* p1_w  = (const float*)d_in[13];
  const float* p1_b  = (const float*)d_in[14];
  const float* p2w0  = (const float*)d_in[15];
  const float* p2b0  = (const float*)d_in[16];
  const float* p2w1  = (const float*)d_in[17];
  const float* p2b1  = (const float*)d_in[18];
  const float* p2w2  = (const float*)d_in[19];
  const float* p2b2  = (const float*)d_in[20];
  const float* p2w3  = (const float*)d_in[21];
  const float* p2b3  = (const float*)d_in[22];

  float* out       = (float*)d_out;
  float* out_probs = out;            // [1024,4]
  float* out_eid   = out + 4096;     // [1024] (stored as float)
  float* out_lat   = out + 5120;     // [1024,120,4,4]

  char* w = (char*)d_ws;
  int*    eid    = (int*)(w + 0);
  int*    counts = (int*)(w + 4096);
  int*    lists  = (int*)(w + 8192);
  float*  feat32 = (float*)(w + 32768);
  half_t* feat16 = (half_t*)(w + 3276800);
  half_t* hbuf   = (half_t*)(w + 4915200);
  half_t* l1buf  = (half_t*)(w + 7340032);     // 32 MB, ends ~40.9 MB
  half_t* p1h    = (half_t*)(w + 41943040ull); // 33.5 MB f16 p1_w
  half_t* p2h    = (half_t*)(w + 75497472ull); // 62.9 MB f16 p2_w (all experts)
  const bool wh = ws_size >= 138412032ull;     // enough scratch for f16 weights

  k_unfold  <<<dim3(1024),  dim3(256),  0, stream>>>(image, feat32, feat16);
  k_router  <<<dim3(1024),  dim3(256),  0, stream>>>(feat32, rw1, rb1, rw2, rb2,
                                                     out_probs, out_eid, eid);
  k_compact <<<dim3(1),     dim3(1024), 0, stream>>>(eid, counts, lists);
  k_zero    <<<dim3(1024),  dim3(256),  0, stream>>>(out_lat, (long)1024 * 1920);

  if (wh) {  // one-time f32->f16 weight conversion (streamed)
    k_cvt_f32_f16<<<dim3(2048), dim3(256), 0, stream>>>(p1_w, p1h, 16777216L / 8);
    k_cvt_f32_f16<<<dim3(256),  dim3(256), 0, stream>>>(p2w0, p2h + 0L,         262144L);
    k_cvt_f32_f16<<<dim3(512),  dim3(256), 0, stream>>>(p2w1, p2h + 2097152L,   524288L);
    k_cvt_f32_f16<<<dim3(1024), dim3(256), 0, stream>>>(p2w2, p2h + 6291456L,  1048576L);
    k_cvt_f32_f16<<<dim3(2048), dim3(256), 0, stream>>>(p2w3, p2h + 14680064L, 2097152L);
  }

  k_frontend<<<dim3(4, 64), dim3(256), 0, stream>>>(feat16, counts, lists,
                                                    pe_w, pe_b, inw, inb, oww, ob,
                                                    ln_g, ln_b, hbuf);
  if (wh) {
    k_p1<true> <<<dim3(4, 16), dim3(256), 0, stream>>>(hbuf, counts, lists,
                                                       p1_w, p1h, p1_b, l1buf);
    k_p2<true> <<<dim3(4, 16), dim3(256), 0, stream>>>(l1buf, counts, lists,
                                                       p2w0, p2w1, p2w2, p2w3, p2h,
                                                       p2b0, p2b1, p2b2, p2b3, out_lat);
  } else {
    k_p1<false><<<dim3(4, 16), dim3(256), 0, stream>>>(hbuf, counts, lists,
                                                       p1_w, p1h, p1_b, l1buf);
    k_p2<false><<<dim3(4, 16), dim3(256), 0, stream>>>(l1buf, counts, lists,
                                                       p2w0, p2w1, p2w2, p2w3, p2h,
                                                       p2b0, p2b1, p2b2, p2b3, out_lat);
  }
}
